// GCN_Direct_89043261980692
// MI455X (gfx1250) — matrix operand order repaired and checked by reading, not compiled, not run on hardware
//
#include <hip/hip_runtime.h>
#include <hip/hip_bf16.h>

// ---------------------------------------------------------------------------
// GCN (3-layer) + dot-product link decode for MI455X (gfx1250).
// GEMMs: v_wmma_f32_16x16x32_bf16 with pre-converted bf16 activations and
// pre-packed B fragments (zero conversion work in the WMMA hot loop).
// Graph scatter: f32 global atomics (agg + hW buffers fit in 192MB L2).
// ---------------------------------------------------------------------------

#define N_NODES 100000
#define N_EDGES 1600000
#define N_PAIRS 200000
#define C_IN    128
#define C_HID   128
#define C_OUT   64

typedef __attribute__((ext_vector_type(16))) __bf16 v16bf;
typedef __attribute__((ext_vector_type(8)))  float  v8f;

struct bf16x16 { uint4 lo; uint4 hi; };   // 32 bytes == v16bf

// Round-to-nearest-even float -> bf16 bits
static __device__ __forceinline__ unsigned short bfbits(float f) {
    unsigned u = __builtin_bit_cast(unsigned, f);
    return (unsigned short)((u + 0x7FFFu + ((u >> 16) & 1u)) >> 16);
}

// ---------------------------------------------------------------------------
__global__ void fill_f32(float* __restrict__ p, float v, int n) {
    int i = blockIdx.x * blockDim.x + threadIdx.x;
    if (i < n) p[i] = v;
}

// f32 -> bf16, 4 elements per thread (vectorized b128 load / b64 store)
__global__ void cvt_f32_bf16(const float* __restrict__ in,
                             unsigned short* __restrict__ out_, int n4) {
    int i = blockIdx.x * blockDim.x + threadIdx.x;
    if (i >= n4) return;
    const float4 v = ((const float4*)in)[i];
    ushort4 o;
    o.x = bfbits(v.x); o.y = bfbits(v.y); o.z = bfbits(v.z); o.w = bfbits(v.w);
    ((ushort4*)out_)[i] = o;
}

// Pack W[Cin x Cout] (f32 row-major) into per-wave WMMA B fragments (bf16):
// Wpack[((tn*KS + ks)*32 + lane)*16 + t] = bf16(W[K*Cout + N])
//   with K = ks*32 + (t>>3)*16 + (lane>>4)*8 + (t&7), N = tn*16 + (lane&15)
__global__ void pack_W_bf16(const float* __restrict__ W,
                            unsigned short* __restrict__ Wpack,
                            int Cout, int KS, int total) {
    int tid = blockIdx.x * blockDim.x + threadIdx.x;
    if (tid >= total) return;
    int t    = tid & 15;
    int lane = (tid >> 4) & 31;
    int ks   = (tid >> 9) % KS;
    int tn   = tid / (512 * KS);
    int K = ks * 32 + (t >> 3) * 16 + (lane >> 4) * 8 + (t & 7);
    int N = tn * 16 + (lane & 15);
    Wpack[tid] = bfbits(W[(size_t)K * Cout + N]);
}

// deg[dst] += 1  (deg pre-filled with 1.0 for the self loop)
__global__ void degree_kernel(const long long* __restrict__ dst,
                              float* __restrict__ deg, int E) {
    int e = blockIdx.x * blockDim.x + threadIdx.x;
    if (e < E) atomicAdd(&deg[(int)dst[e]], 1.0f);
}

__global__ void node_norm_kernel(const float* __restrict__ deg,
                                 float* __restrict__ dinv_sqrt,
                                 float* __restrict__ deg_inv, int n) {
    int i = blockIdx.x * blockDim.x + threadIdx.x;
    if (i < n) {
        float d = deg[i];
        dinv_sqrt[i] = rsqrtf(d);
        deg_inv[i]   = 1.0f / d;
    }
}

__global__ void edge_w_kernel(const long long* __restrict__ src,
                              const long long* __restrict__ dst,
                              const float* __restrict__ dinv_sqrt,
                              float* __restrict__ w_e, int E) {
    int e = blockIdx.x * blockDim.x + threadIdx.x;
    if (e < E) w_e[e] = dinv_sqrt[(int)src[e]] * dinv_sqrt[(int)dst[e]];
}

// ---------------------------------------------------------------------------
// Hout[N x Cout] = Hbf[N x CIN] @ W  (W pre-packed as B fragments)
// One wave per 16x16 output tile; CIN templated so the K loop fully unrolls
// into CIN/32 back-to-back v_wmma_f32_16x16x32_bf16.
// All control flow is wave-uniform (EXEC all-1s, required by WMMA).
// ---------------------------------------------------------------------------
template <int CIN>
__global__ void gemm_wmma_bf16(const unsigned short* __restrict__ Hbf,
                               const unsigned short* __restrict__ Wpack,
                               float* __restrict__ Hout,
                               int Cout, int n_mtiles) {
    constexpr int KS = CIN / 32;
    const int wave   = threadIdx.x >> 5;
    const int lane   = threadIdx.x & 31;
    const int tile_m = blockIdx.x * 8 + wave;      // uniform within wave
    const int tile_n = blockIdx.y;
    if (tile_m >= n_mtiles) return;                // wave-uniform exit

    const int l15 = lane & 15;
    const int khi = lane >> 4;                     // K-half select
    const int row = tile_m * 16 + l15;             // A row (M)
    const int col = tile_n * 16 + l15;             // B/C col (N)

    v8f acc = {};
#pragma unroll
    for (int ks = 0; ks < KS; ++ks) {
        // A fragment (16-bit A 16x32 layout): two contiguous 16B runs
        const unsigned short* ap = Hbf + (size_t)row * CIN + ks * 32 + khi * 8;
        bf16x16 at;
        at.lo = *(const uint4*)(ap);        // K = kk .. kk+7   (this half)
        at.hi = *(const uint4*)(ap + 16);   // K = kk+16 .. kk+23
        v16bf a = __builtin_bit_cast(v16bf, at);

        // B fragment: pre-packed, 32 contiguous bytes per lane
        const bf16x16* bp =
            (const bf16x16*)Wpack + ((size_t)tile_n * KS + ks) * 32 + lane;
        v16bf b = __builtin_bit_cast(v16bf, *bp);

        acc = __builtin_amdgcn_wmma_f32_16x16x32_bf16(
            /*neg_a=*/false, a, /*neg_b=*/false, b,
            /*c_mod=*/(short)0, acc, /*reuse_a=*/false, /*reuse_b=*/false);
    }

    // C/D layout: VGPR v -> M = v + 8*khi, N = col
#pragma unroll
    for (int v = 0; v < 8; ++v) {
        int m = tile_m * 16 + v + 8 * khi;
        Hout[(size_t)m * Cout + col] = acc[v];
    }
}

// ---------------------------------------------------------------------------
// agg[dst[e], :] += hW[src[e], :] * w_e[e]
// One thread per (edge, 4-channel group): float4 gather + 4 f32 atomics (L2).
__global__ void scatter_kernel(const float* __restrict__ hW,
                               const float* __restrict__ w_e,
                               const long long* __restrict__ src,
                               const long long* __restrict__ dst,
                               float* __restrict__ agg,
                               int C, int qshift, int total) {
    int tid = blockIdx.x * blockDim.x + threadIdx.x;
    if (tid >= total) return;
    int e = tid >> qshift;
    int q = tid & ((1 << qshift) - 1);
    float w = w_e[e];
    int s = (int)src[e];
    int d = (int)dst[e];
    const float4 v = *(const float4*)(hW + (size_t)s * C + q * 4);
    float* ap = agg + (size_t)d * C + q * 4;
    atomicAdd(ap + 0, v.x * w);
    atomicAdd(ap + 1, v.y * w);
    atomicAdd(ap + 2, v.z * w);
    atomicAdd(ap + 3, v.w * w);
}

// hW[n,c] = (relu?) (agg[n,c] + hW[n,c]*deg_inv[n] + bias[c])   (in place)
__global__ void combine_kernel(float* __restrict__ hW,
                               const float* __restrict__ agg,
                               const float* __restrict__ deg_inv,
                               const float* __restrict__ bias,
                               int C, int do_relu, int total) {
    int i = blockIdx.x * blockDim.x + threadIdx.x;
    if (i >= total) return;
    int n = i / C;
    int c = i - n * C;
    float v = agg[i] + hW[i] * deg_inv[n] + bias[c];
    if (do_relu) v = fmaxf(v, 0.0f);
    hW[i] = v;
}

// out[p] = dot(h[node_i[p], :64], h[node_j[p], :64])  — one wave per pair
__global__ void decode_kernel(const float* __restrict__ h,
                              const long long* __restrict__ ni,
                              const long long* __restrict__ nj,
                              float* __restrict__ out, int P) {
    int wid  = (blockIdx.x * blockDim.x + threadIdx.x) >> 5; // uniform per wave
    int lane = threadIdx.x & 31;
    if (wid >= P) return;
    long long i = ni[wid];
    long long j = nj[wid];
    const float2 a = *(const float2*)(h + (size_t)i * C_OUT + lane * 2);
    const float2 b = *(const float2*)(h + (size_t)j * C_OUT + lane * 2);
    float s = a.x * b.x + a.y * b.y;
#pragma unroll
    for (int o = 16; o > 0; o >>= 1) s += __shfl_xor(s, o, 32);
    if (lane == 0) out[wid] = s;
}

// ---------------------------------------------------------------------------
extern "C" void kernel_launch(void* const* d_in, const int* in_sizes, int n_in,
                              void* d_out, int out_size, void* d_ws, size_t ws_size,
                              hipStream_t stream) {
    const float*     x   = (const float*)d_in[0];
    const long long* ei  = (const long long*)d_in[1];   // [2, E] int64
    const long long* ni  = (const long long*)d_in[2];
    const long long* nj  = (const long long*)d_in[3];
    const float*     W1  = (const float*)d_in[4];
    const float*     b1  = (const float*)d_in[5];
    const float*     W2  = (const float*)d_in[6];
    const float*     b2  = (const float*)d_in[7];
    const float*     W3  = (const float*)d_in[8];
    const float*     b3  = (const float*)d_in[9];
    float*           out = (float*)d_out;

    const long long* src = ei;            // row 0
    const long long* dst = ei + N_EDGES;  // row 1

    // workspace carve-out (256B aligned slices)
    char*  base = (char*)d_ws;
    size_t off  = 0;
    auto alloc = [&](size_t bytes) -> char* {
        char* p = base + off;
        off = (off + bytes + 255) & ~(size_t)255;
        return p;
    };
    float*          deg       = (float*)alloc((size_t)N_NODES * 4);
    float*          dinv_sqrt = (float*)alloc((size_t)N_NODES * 4);
    float*          deg_inv   = (float*)alloc((size_t)N_NODES * 4);
    float*          w_e       = (float*)alloc((size_t)N_EDGES * 4);
    float*          bufA      = (float*)alloc((size_t)N_NODES * C_HID * 4); // 51.2 MB
    float*          bufB      = (float*)alloc((size_t)N_NODES * C_HID * 4); // 51.2 MB
    unsigned short* Hbf       = (unsigned short*)alloc((size_t)N_NODES * C_HID * 2); // 25.6 MB
    unsigned short* Wpack     = (unsigned short*)alloc((size_t)(C_HID / 16) * (C_HID / 32) * 512 * 2);

    const int BT = 256;
    auto blocks = [](long long n, int bt) { return (int)((n + bt - 1) / bt); };

    // ---- normalization terms ----
    fill_f32<<<blocks(N_NODES, BT), BT, 0, stream>>>(deg, 1.0f, N_NODES); // self loop
    degree_kernel<<<blocks(N_EDGES, BT), BT, 0, stream>>>(dst, deg, N_EDGES);
    node_norm_kernel<<<blocks(N_NODES, BT), BT, 0, stream>>>(deg, dinv_sqrt, deg_inv, N_NODES);
    edge_w_kernel<<<blocks(N_EDGES, BT), BT, 0, stream>>>(src, dst, dinv_sqrt, w_e, N_EDGES);

    const int MT = N_NODES / 16;   // 6250 M-tiles (exact)
    const int KS = C_IN / 32;      // 4 K-steps

    // ---- layer 1: x -> bufA ----
    {
        int n4 = N_NODES * C_IN / 4;
        cvt_f32_bf16<<<blocks(n4, BT), BT, 0, stream>>>(x, Hbf, n4);
        int wtotal = (C_HID / 16) * KS * 512;
        pack_W_bf16<<<blocks(wtotal, BT), BT, 0, stream>>>(W1, Wpack, C_HID, KS, wtotal);
        dim3 g(blocks(MT, 8), C_HID / 16);
        gemm_wmma_bf16<C_IN><<<g, 256, 0, stream>>>(Hbf, Wpack, bufA, C_HID, MT);
        int nc = N_NODES * C_HID;
        fill_f32<<<blocks(nc, BT), BT, 0, stream>>>(bufB, 0.0f, nc);
        int total = N_EDGES * (C_HID / 4);
        scatter_kernel<<<blocks(total, BT), BT, 0, stream>>>(bufA, w_e, src, dst, bufB,
                                                             C_HID, 5, total);
        combine_kernel<<<blocks(nc, BT), BT, 0, stream>>>(bufA, bufB, deg_inv, b1,
                                                          C_HID, 1, nc);
    }
    // ---- layer 2: bufA -> bufB ----
    {
        int n4 = N_NODES * C_HID / 4;
        cvt_f32_bf16<<<blocks(n4, BT), BT, 0, stream>>>(bufA, Hbf, n4);
        int wtotal = (C_HID / 16) * KS * 512;
        pack_W_bf16<<<blocks(wtotal, BT), BT, 0, stream>>>(W2, Wpack, C_HID, KS, wtotal);
        dim3 g(blocks(MT, 8), C_HID / 16);
        gemm_wmma_bf16<C_HID><<<g, 256, 0, stream>>>(Hbf, Wpack, bufB, C_HID, MT);
        int nc = N_NODES * C_HID;
        fill_f32<<<blocks(nc, BT), BT, 0, stream>>>(bufA, 0.0f, nc);
        int total = N_EDGES * (C_HID / 4);
        scatter_kernel<<<blocks(total, BT), BT, 0, stream>>>(bufB, w_e, src, dst, bufA,
                                                             C_HID, 5, total);
        combine_kernel<<<blocks(nc, BT), BT, 0, stream>>>(bufB, bufA, deg_inv, b2,
                                                          C_HID, 1, nc);
    }
    // ---- layer 3: bufB -> bufA (C_OUT=64, no relu) ----
    {
        int n4 = N_NODES * C_HID / 4;
        cvt_f32_bf16<<<blocks(n4, BT), BT, 0, stream>>>(bufB, Hbf, n4);
        int wtotal = (C_OUT / 16) * KS * 512;
        pack_W_bf16<<<blocks(wtotal, BT), BT, 0, stream>>>(W3, Wpack, C_OUT, KS, wtotal);
        dim3 g(blocks(MT, 8), C_OUT / 16);
        gemm_wmma_bf16<C_HID><<<g, 256, 0, stream>>>(Hbf, Wpack, bufA, C_OUT, MT);
        int nc = N_NODES * C_OUT;
        fill_f32<<<blocks(nc, BT), BT, 0, stream>>>(bufB, 0.0f, nc);
        int total = N_EDGES * (C_OUT / 4);
        scatter_kernel<<<blocks(total, BT), BT, 0, stream>>>(bufA, w_e, src, dst, bufB,
                                                             C_OUT, 4, total);
        combine_kernel<<<blocks(nc, BT), BT, 0, stream>>>(bufA, bufB, deg_inv, b3,
                                                          C_OUT, 0, nc);
    }

    // ---- pair decode: one wave per pair ----
    {
        long long threads = (long long)N_PAIRS * 32;
        decode_kernel<<<blocks(threads, BT), BT, 0, stream>>>(bufA, ni, nj, out, N_PAIRS);
    }
}
